// Model_7456063226099
// MI455X (gfx1250) — matrix operand order repaired and checked by reading, not compiled, not run on hardware
//
#include <hip/hip_runtime.h>
#include <hip/hip_bf16.h>
#include <math.h>

// ---------------- constants from the reference ----------------
#define NPTS_D  2
#define NPTS_C  1
#define LL1     16
#define LL2     32
#define LATF    64
#define LL3     32
#define DEF     5
#define EF      11          // DE*D+1
#define KNBR    16
#define RESV    64
#define SNAPC   7
#define OUTF    6

// ---------------- input pointer indices (setup_inputs dict order) ------------
enum {
  IN_MEANS = 0, IN_COV, IN_U, IN_SNAP, IN_NBR,
  P_SP_W1, P_SP_B1, P_SP_W2, P_SP_B2, P_SP_W3, P_SP_B3,
  P_LT_W1, P_LT_B1, P_LT_W2, P_LT_B2, P_LT_W3, P_LT_B3,
  P_TN_W1, P_TN_B1, P_TN_W2, P_TN_B2, P_TN_W3, P_TN_B3,
  P_TV_W1, P_TV_B1, P_TV_W2, P_TV_B2, P_TV_W3, P_TV_B3,
  P_IP_W1, P_IP_B1, P_IP_W2, P_IP_B2, P_IP_W3, P_IP_B3,
  P_DT_W1, P_DT_B1, P_DT_W2, P_DT_B2,
  P_DN_W1, P_DN_B1, P_DN_W2, P_DN_B2, P_DN_W3, P_DN_B3
};

// ---------------- workspace layout (float offsets) ----------------
#define WS_CONV1  0u                       // 16*62*62 = 61504
#define WS_CONV2  61504u                   // 32*60*60 = 115200
#define WS_CONV3  176704u                  // 64*58*58 = 215296
#define WS_SNAPF  392000u                  // 64
#define WS_HSUM   392064u                  // 64
#define WS_TT     392128u                  // 5 (T 2x2 + Tv)
#define WS_GF     392192u                  // n*64
// fA = WS_GF + n*64 ; fB = fA + n*64   (computed at launch)

// ---------------- fast activations (native V_TANH_F32 on gfx1250) ------------
__device__ __forceinline__ float fast_tanh(float x) {
#if __has_builtin(__builtin_amdgcn_tanhf)
  return __builtin_amdgcn_tanhf(x);   // v_tanh_f32 (TRANS32; co-executes with WMMA)
#else
  return tanhf(x);
#endif
}

// ---------------- WMMA helpers (CDNA5 gfx1250, wave32) ----------------
typedef __attribute__((ext_vector_type(16))) _Float16 v16h;
typedef __attribute__((ext_vector_type(8)))  float    v8f;

__device__ __forceinline__ v8f wmma16(v16h a, v16h b, v8f c) {
  // D = A(16x32 f16) x B(32x16 f16) + C(16x16 f32)
  return __builtin_amdgcn_wmma_f32_16x16x32_f16(
      /*neg_a=*/false, a, /*neg_b=*/false, b,
      /*c_mod=*/(short)0, c, /*reuse_a=*/false, /*reuse_b=*/false);
}

// A-fragment (16x32 f16 tile, M=row) from a row-major f16 LDS array.
// Lanes 0-15: row M=lane, halves 0..15 => K=kbase+0..15
// Lanes 16-31: row M=lane-16, halves 0..15 => K=kbase+16..31 (matches frag_b_wglb)
__device__ __forceinline__ v16h frag_a_lds(const _Float16* s, int ld, int kbase, int lane) {
  const int m = lane & 15;
  const int h = (lane >> 4) << 4;
  const _Float16* row = s + m * ld + kbase;
  v16h a;
#pragma unroll
  for (int i = 0; i < 8; ++i) { a[i] = row[h + i]; a[8 + i] = row[h + 8 + i]; }
  return a;
}

// B-fragment (32x16 f16 tile, B[k][n] = W[n][k]) read from a row-major fp32
// weight matrix W[out=n][in=k], converting to f16 in-flight. All call sites are
// 64B-aligned (ldk multiple of 16, nbase/kbase multiples of 16) -> b128 loads.
// Lanes 0-15: N=lane, halves = K kbase+0..15 ; lanes 16-31: N=lane-16, K kbase+16..31
__device__ __forceinline__ v16h frag_b_wglb(const float* __restrict__ w, int ldk,
                                            int nbase, int kbase, int lane) {
  const float4* row = (const float4*)(w + (size_t)(nbase + (lane & 15)) * ldk
                                        + kbase + ((lane >> 4) << 4));
  v16h b;
#pragma unroll
  for (int q = 0; q < 4; ++q) {
    float4 v = row[q];
    b[q * 4 + 0] = (_Float16)v.x;
    b[q * 4 + 1] = (_Float16)v.y;
    b[q * 4 + 2] = (_Float16)v.z;
    b[q * 4 + 3] = (_Float16)v.w;
  }
  return b;
}

// ---------------- tiny utility kernels ----------------
__global__ void k_zero(float* __restrict__ p, int n) {
  int i = blockIdx.x * blockDim.x + threadIdx.x;
  if (i < n) p[i] = 0.f;
}

// Generic 3x3 VALID conv + tanh. Input strides allow HWC (snapshots) or CHW (ws).
__global__ void k_conv(const float* __restrict__ in, const float* __restrict__ w,
                       const float* __restrict__ b, float* __restrict__ out,
                       int Cin, int Cout, int Hout, int Wout,
                       int ics, int iys, int ixs) {
  int idx = blockIdx.x * blockDim.x + threadIdx.x;
  int total = Cout * Hout * Wout;
  if (idx >= total) return;
  int co = idx / (Hout * Wout);
  int rem = idx - co * Hout * Wout;
  int y = rem / Wout, x = rem - y * Wout;
  float s = b[co];
  for (int ci = 0; ci < Cin; ++ci) {
    const float* wr = w + ((co * Cin + ci) * 9);
#pragma unroll
    for (int ky = 0; ky < 3; ++ky)
#pragma unroll
      for (int kx = 0; kx < 3; ++kx)
        s += wr[ky * 3 + kx] * in[ci * ics + (y + ky) * iys + (x + kx) * ixs];
  }
  out[idx] = fast_tanh(s);
}

// per-channel spatial mean of conv3 output -> snap_feat[64]
__global__ void k_snap_mean(const float* __restrict__ c3, float* __restrict__ snap) {
  __shared__ float red[256];
  const int c = blockIdx.x;
  const int HW = 58 * 58;
  float s = 0.f;
  for (int i = threadIdx.x; i < HW; i += 256) s += c3[c * HW + i];
  red[threadIdx.x] = s;
  __syncthreads();
  for (int o = 128; o > 0; o >>= 1) {
    if (threadIdx.x < (unsigned)o) red[threadIdx.x] += red[threadIdx.x + o];
    __syncthreads();
  }
  if (threadIdx.x == 0) snap[c] = red[0] / (float)HW;
}

// InputTransform latent MLP (7->16->32->64, tanh all) + global sum of h3
__global__ void k_latent(const float* __restrict__ means, const float* __restrict__ cov,
                         const float* __restrict__ u,
                         const float* __restrict__ w1, const float* __restrict__ b1,
                         const float* __restrict__ w2, const float* __restrict__ b2,
                         const float* __restrict__ w3, const float* __restrict__ b3,
                         float* __restrict__ hsum, int n) {
  __shared__ float acc[LATF];
  const int t = threadIdx.x;
  if (t < LATF) acc[t] = 0.f;
  __syncthreads();
  const int p = blockIdx.x * blockDim.x + t;
  if (p < n) {
    float par[7] = { means[p * 2], means[p * 2 + 1],
                     cov[p * 4], cov[p * 4 + 1], cov[p * 4 + 2], cov[p * 4 + 3],
                     u[p] };
    float h1[LL1], h2[LL2];
#pragma unroll
    for (int j = 0; j < LL1; ++j) {
      float s = b1[j];
#pragma unroll
      for (int i = 0; i < 7; ++i) s += par[i] * w1[j * 7 + i];
      h1[j] = fast_tanh(s);
    }
#pragma unroll
    for (int j = 0; j < LL2; ++j) {
      float s = b2[j];
#pragma unroll
      for (int i = 0; i < LL1; ++i) s += h1[i] * w2[j * LL1 + i];
      h2[j] = fast_tanh(s);
    }
    for (int j = 0; j < LATF; ++j) {
      float s = b3[j];
#pragma unroll
      for (int i = 0; i < LL2; ++i) s += h2[i] * w3[j * LL2 + i];
      atomicAdd(&acc[j], fast_tanh(s));
    }
  }
  __syncthreads();
  if (t < LATF) atomicAdd(&hsum[t], acc[t]);
}

// latent -> T (2x2, eye + MLP) and Tv (1x1, eye + MLP); single 64-thread block
__global__ void k_transform(const float* __restrict__ hsum, const float* __restrict__ snap,
                            const float* __restrict__ tn_w1, const float* __restrict__ tn_b1,
                            const float* __restrict__ tn_w2, const float* __restrict__ tn_b2,
                            const float* __restrict__ tn_w3, const float* __restrict__ tn_b3,
                            const float* __restrict__ tv_w1, const float* __restrict__ tv_b1,
                            const float* __restrict__ tv_w2, const float* __restrict__ tv_b2,
                            const float* __restrict__ tv_w3, const float* __restrict__ tv_b3,
                            float* __restrict__ TT, int n) {
  __shared__ float lat[2 * LATF], a1[64], a2[64];
  const int t = threadIdx.x;
  lat[t] = hsum[t] / (float)n;
  lat[64 + t] = snap[t];
  __syncthreads();
  if (t < 32) {
    float s = tn_b1[t];
    for (int i = 0; i < 128; ++i) s += lat[i] * tn_w1[t * 128 + i];
    a1[t] = fast_tanh(s);
  } else {
    int j = t - 32;
    float s = tv_b1[j];
    for (int i = 0; i < 128; ++i) s += lat[i] * tv_w1[j * 128 + i];
    a1[t] = fast_tanh(s);
  }
  __syncthreads();
  if (t < 32) {
    float s = tn_b2[t];
    for (int i = 0; i < 32; ++i) s += a1[i] * tn_w2[t * 32 + i];
    a2[t] = fast_tanh(s);
  } else {
    int j = t - 32;
    float s = tv_b2[j];
    for (int i = 0; i < 32; ++i) s += a1[32 + i] * tv_w2[j * 32 + i];
    a2[t] = fast_tanh(s);
  }
  __syncthreads();
  if (t < 4) {
    float s = tn_b3[t];
    for (int i = 0; i < 32; ++i) s += a2[i] * tn_w3[t * 32 + i];
    TT[t] = ((t == 0 || t == 3) ? 1.f : 0.f) + s;
  }
  if (t == 4) {
    float s = tv_b3[0];
    for (int i = 0; i < 32; ++i) s += a2[32 + i] * tv_w3[i];
    TT[4] = 1.f + s;
  }
}

// t_cov/t_u then input_projection MLP (5->16 tanh ->32 tanh ->64 linear) -> gf
__global__ void k_gf(const float* __restrict__ cov, const float* __restrict__ u,
                     const float* __restrict__ TT,
                     const float* __restrict__ w1, const float* __restrict__ b1,
                     const float* __restrict__ w2, const float* __restrict__ b2,
                     const float* __restrict__ w3, const float* __restrict__ b3,
                     float* __restrict__ gf, int n) {
  const int p = blockIdx.x * blockDim.x + threadIdx.x;
  if (p >= n) return;
  const float T00 = TT[0], T01 = TT[1], T10 = TT[2], T11 = TT[3], Tv = TT[4];
  const float c00 = cov[p * 4], c01 = cov[p * 4 + 1], c10 = cov[p * 4 + 2], c11 = cov[p * 4 + 3];
  float tp[5] = { T00 * c00 + T01 * c10, T00 * c01 + T01 * c11,
                  T10 * c00 + T11 * c10, T10 * c01 + T11 * c11,
                  Tv * u[p] };
  float h1[LL1], h2[LL2];
#pragma unroll
  for (int j = 0; j < LL1; ++j) {
    float s = b1[j];
#pragma unroll
    for (int i = 0; i < 5; ++i) s += tp[i] * w1[j * 5 + i];
    h1[j] = fast_tanh(s);
  }
#pragma unroll
  for (int j = 0; j < LL2; ++j) {
    float s = b2[j];
#pragma unroll
    for (int i = 0; i < LL1; ++i) s += h1[i] * w2[j * LL1 + i];
    h2[j] = fast_tanh(s);
  }
  float* dst = gf + (size_t)p * LATF;
  for (int j = 0; j < LATF; ++j) {
    float s = b3[j];
#pragma unroll
    for (int i = 0; i < LL2; ++i) s += h2[i] * w3[j * LL2 + i];
    dst[j] = s;  // no activation on ip layer 3
  }
}

// One aggregation round, fully fused:
//   dh = tanh(feats @ dt_w1^T + b1)              (WMMA)
//   W  = dh @ dt_w2^T + b2   [16 x 704]          (WMMA, never leaves registers)
//   feats_new[p,l] = (1/K) * sum_e W[p,l,e] * M[p,e,l]
//   with M[p,e,l] = sum_k emb[p,k,e] * featsN[p,k,l]  (computed on the fly)
// Block = 128 threads (4 waves) per 16-point tile; waves split the 44 N-chunks.
__global__ void __launch_bounds__(128)
k_agg(const float* __restrict__ fin, float* __restrict__ fout,
      const float* __restrict__ means, const int* __restrict__ nbr,
      const float* __restrict__ w1, const float* __restrict__ b1,
      const float* __restrict__ w2, const float* __restrict__ b2, int n) {
  __shared__ _Float16 sA[16][LATF];            //  2 KB  feats tile (f16)
  __shared__ _Float16 sFN[16][KNBR][LATF];     // 32 KB  gathered neighbor feats (f16)
  __shared__ float    sEmb[16][KNBR][12];      // 12 KB  sin distance embedding (E=11 padded)
  __shared__ _Float16 sDH[16][LATF];           //  2 KB  tanh hidden (f16)
  __shared__ float    sOut[16][LATF];          //  4 KB  output accumulator

  const int t = threadIdx.x;
  const int lane = t & 31;
  const int wave = t >> 5;
  const int p0 = blockIdx.x * 16;

  // stage feats tile + zero accumulators
  for (int i = t; i < 16 * LATF; i += 128) {
    int pp = i >> 6, l = i & 63;
    sA[pp][l] = (_Float16)fin[(size_t)(p0 + pp) * LATF + l];
    sOut[pp][l] = 0.f;
  }
  // gather neighbors: 256 (p,k) pairs; embedding + f16 neighbor feats
  for (int pair = t; pair < 16 * KNBR; pair += 128) {
    int pp = pair >> 4, kk = pair & 15;
    int idx = nbr[(size_t)(p0 + pp) * KNBR + kk];
    float dx = means[(size_t)idx * 2]     - means[(size_t)(p0 + pp) * 2];
    float dy = means[(size_t)idx * 2 + 1] - means[(size_t)(p0 + pp) * 2 + 1];
    sEmb[pp][kk][0] = 1.f;
    float fr = 1.f;
#pragma unroll
    for (int f = 0; f < DEF; ++f) {
      sEmb[pp][kk][1 + f * 2]     = __sinf(dx * fr);
      sEmb[pp][kk][1 + f * 2 + 1] = __sinf(dy * fr);
      fr *= 2.f;
    }
    const float4* src = (const float4*)(fin + (size_t)idx * LATF);
#pragma unroll
    for (int q = 0; q < 16; ++q) {
      float4 v = src[q];
      sFN[pp][kk][q * 4 + 0] = (_Float16)v.x;
      sFN[pp][kk][q * 4 + 1] = (_Float16)v.y;
      sFN[pp][kk][q * 4 + 2] = (_Float16)v.z;
      sFN[pp][kk][q * 4 + 3] = (_Float16)v.w;
    }
  }
  __syncthreads();

  // Phase 1: dh = tanh(feats @ W1^T + b1). Wave w owns output cols [16w, 16w+16).
  {
    v16h a0 = frag_a_lds(&sA[0][0], LATF, 0, lane);
    v16h a1 = frag_a_lds(&sA[0][0], LATF, 32, lane);
    const int nb = wave * 16;
    v16h bf0 = frag_b_wglb(w1, LATF, nb, 0, lane);
    v16h bf1 = frag_b_wglb(w1, LATF, nb, 32, lane);
    v8f c = {};
    c = wmma16(a0, bf0, c);
    c = wmma16(a1, bf1, c);
    const int nn = nb + (lane & 15);
    const int mh = (lane >> 4) * 8;
    const float bias = b1[nn];
#pragma unroll
    for (int r = 0; r < 8; ++r) sDH[mh + r][nn] = (_Float16)fast_tanh(c[r] + bias);
  }
  __syncthreads();

  // Phase 2: 44 chunks of 16 cols of W = dh @ W2^T; fuse with aggregation.
  {
    v16h a0 = frag_a_lds(&sDH[0][0], LATF, 0, lane);
    v16h a1 = frag_a_lds(&sDH[0][0], LATF, 32, lane);
    const int mh = (lane >> 4) * 8;
    for (int ch = wave; ch < (EF * LATF) / 16; ch += 4) {
      const int colb = ch * 16;
      v16h bf0 = frag_b_wglb(w2, LATF, colb, 0, lane);
      v16h bf1 = frag_b_wglb(w2, LATF, colb, 32, lane);
      v8f c = {};
      c = wmma16(a0, bf0, c);
      c = wmma16(a1, bf1, c);
      const int col = colb + (lane & 15);
      const int l = col / EF;
      const int e = col - l * EF;
      const float bias = b2[col];
#pragma unroll
      for (int r = 0; r < 8; ++r) {
        const int pp = mh + r;
        float m = 0.f;
#pragma unroll
        for (int kk = 0; kk < KNBR; ++kk)
          m += sEmb[pp][kk][e] * (float)sFN[pp][kk][l];
        atomicAdd(&sOut[pp][l], (c[r] + bias) * m);
      }
    }
  }
  __syncthreads();
  for (int i = t; i < 16 * LATF; i += 128) {
    int pp = i >> 6, l = i & 63;
    fout[(size_t)(p0 + pp) * LATF + l] = sOut[pp][l] * (1.f / (float)KNBR);
  }
}

// Delta network: lg=[gf|feats|snap] (192) ->64 tanh ->32 tanh ->6 linear
__global__ void __launch_bounds__(128)
k_delta(const float* __restrict__ gf, const float* __restrict__ feats,
        const float* __restrict__ snap,
        const float* __restrict__ w1, const float* __restrict__ b1,
        const float* __restrict__ w2, const float* __restrict__ b2,
        const float* __restrict__ w3, const float* __restrict__ b3,
        float* __restrict__ out, int n) {
  __shared__ _Float16 sLG[16][3 * LATF];   // 6 KB
  __shared__ _Float16 sH1[16][LATF];       // 2 KB
  __shared__ float    sH2[16][LL2];        // 2 KB
  const int t = threadIdx.x;
  const int lane = t & 31;
  const int wave = t >> 5;
  const int p0 = blockIdx.x * 16;

  for (int i = t; i < 16 * 3 * LATF; i += 128) {
    int pp = i / (3 * LATF), k = i - pp * (3 * LATF);
    float v = (k < LATF) ? gf[(size_t)(p0 + pp) * LATF + k]
            : (k < 2 * LATF) ? feats[(size_t)(p0 + pp) * LATF + (k - LATF)]
            : snap[k - 2 * LATF];
    sLG[pp][k] = (_Float16)v;
  }
  __syncthreads();

  // layer 1: K=192 -> 6 WMMA k-steps; wave w owns 16 output cols
  {
    const int nb = wave * 16;
    v8f c = {};
#pragma unroll
    for (int ks = 0; ks < 6; ++ks) {
      v16h a = frag_a_lds(&sLG[0][0], 3 * LATF, ks * 32, lane);
      v16h b = frag_b_wglb(w1, 3 * LATF, nb, ks * 32, lane);
      c = wmma16(a, b, c);
    }
    const int nn = nb + (lane & 15);
    const int mh = (lane >> 4) * 8;
    const float bias = b1[nn];
#pragma unroll
    for (int r = 0; r < 8; ++r) sH1[mh + r][nn] = (_Float16)fast_tanh(c[r] + bias);
  }
  __syncthreads();

  // layer 2: 64 -> 32 (2 chunks on waves 0,1)
  if (wave < 2) {
    const int nb = wave * 16;
    v16h a0 = frag_a_lds(&sH1[0][0], LATF, 0, lane);
    v16h a1 = frag_a_lds(&sH1[0][0], LATF, 32, lane);
    v16h bw0 = frag_b_wglb(w2, LATF, nb, 0, lane);
    v16h bw1 = frag_b_wglb(w2, LATF, nb, 32, lane);
    v8f c = {};
    c = wmma16(a0, bw0, c);
    c = wmma16(a1, bw1, c);
    const int nn = nb + (lane & 15);
    const int mh = (lane >> 4) * 8;
    const float bias = b2[nn];
#pragma unroll
    for (int r = 0; r < 8; ++r) sH2[mh + r][nn] = fast_tanh(c[r] + bias);
  }
  __syncthreads();

  // layer 3: 32 -> 6 (scalar)
  if (t < 16 * OUTF) {
    int pp = t / OUTF, o = t - pp * OUTF;
    float s = b3[o];
#pragma unroll
    for (int j = 0; j < LL2; ++j) s += sH2[pp][j] * w3[o * LL2 + j];
    out[(size_t)(p0 + pp) * OUTF + o] = s;
  }
}

// ---------------- host launcher ----------------
extern "C" void kernel_launch(void* const* d_in, const int* in_sizes, int n_in,
                              void* d_out, int out_size, void* d_ws, size_t ws_size,
                              hipStream_t stream) {
  (void)n_in; (void)out_size; (void)ws_size;
  const int n = in_sizes[IN_MEANS] / NPTS_D;     // 65536

  const float* means = (const float*)d_in[IN_MEANS];
  const float* cov   = (const float*)d_in[IN_COV];
  const float* u     = (const float*)d_in[IN_U];
  const float* snapI = (const float*)d_in[IN_SNAP];   // [RES,RES,SNAP] HWC
  const int*   nbr   = (const int*)d_in[IN_NBR];
  #define PF(i) ((const float*)d_in[(i)])

  float* ws = (float*)d_ws;
  float* c1   = ws + WS_CONV1;
  float* c2   = ws + WS_CONV2;
  float* c3   = ws + WS_CONV3;
  float* snap = ws + WS_SNAPF;
  float* hsum = ws + WS_HSUM;
  float* TT   = ws + WS_TT;
  float* gf   = ws + WS_GF;
  float* fA   = gf + (size_t)n * LATF;
  float* fB   = fA + (size_t)n * LATF;
  float* outp = (float*)d_out;

  // 0) zero the h-mean accumulator (deterministic per call)
  k_zero<<<1, 64, 0, stream>>>(hsum, LATF);

  // 1) snapshot conv stack (tanh each layer)
  {
    int tot1 = LL1 * 62 * 62;
    k_conv<<<(tot1 + 255) / 256, 256, 0, stream>>>(
        snapI, PF(P_SP_W1), PF(P_SP_B1), c1,
        SNAPC, LL1, 62, 62, /*ics*/1, /*iys*/RESV * SNAPC, /*ixs*/SNAPC);
    int tot2 = LL2 * 60 * 60;
    k_conv<<<(tot2 + 255) / 256, 256, 0, stream>>>(
        c1, PF(P_SP_W2), PF(P_SP_B2), c2,
        LL1, LL2, 60, 60, 62 * 62, 62, 1);
    int tot3 = LATF * 58 * 58;
    k_conv<<<(tot3 + 255) / 256, 256, 0, stream>>>(
        c2, PF(P_SP_W3), PF(P_SP_B3), c3,
        LL2, LATF, 58, 58, 60 * 60, 60, 1);
    k_snap_mean<<<LATF, 256, 0, stream>>>(c3, snap);
  }

  // 2) per-point latent MLP + mean accumulation
  k_latent<<<(n + 255) / 256, 256, 0, stream>>>(
      means, cov, u,
      PF(P_LT_W1), PF(P_LT_B1), PF(P_LT_W2), PF(P_LT_B2), PF(P_LT_W3), PF(P_LT_B3),
      hsum, n);

  // 3) latent -> T, Tv
  k_transform<<<1, 64, 0, stream>>>(
      hsum, snap,
      PF(P_TN_W1), PF(P_TN_B1), PF(P_TN_W2), PF(P_TN_B2), PF(P_TN_W3), PF(P_TN_B3),
      PF(P_TV_W1), PF(P_TV_B1), PF(P_TV_W2), PF(P_TV_B2), PF(P_TV_W3), PF(P_TV_B3),
      TT, n);

  // 4) transformed inputs -> per-node global features gf
  k_gf<<<(n + 255) / 256, 256, 0, stream>>>(
      cov, u, TT,
      PF(P_IP_W1), PF(P_IP_B1), PF(P_IP_W2), PF(P_IP_B2), PF(P_IP_W3), PF(P_IP_B3),
      gf, n);

  // 5) two fused WMMA aggregation rounds
  const int tiles = n / 16;
  k_agg<<<tiles, 128, 0, stream>>>(gf, fA, means, nbr,
      PF(P_DT_W1), PF(P_DT_B1), PF(P_DT_W2), PF(P_DT_B2), n);
  k_agg<<<tiles, 128, 0, stream>>>(fA, fB, means, nbr,
      PF(P_DT_W1), PF(P_DT_B1), PF(P_DT_W2), PF(P_DT_B2), n);

  // 6) delta network
  k_delta<<<tiles, 128, 0, stream>>>(gf, fB, snap,
      PF(P_DN_W1), PF(P_DN_B1), PF(P_DN_W2), PF(P_DN_B2), PF(P_DN_W3), PF(P_DN_B3),
      outp, n);
  #undef PF
}
